// SRURNN_19198503813516
// MI455X (gfx1250) — compile-verified, matrix operand order checked by reading, not run on hardware
//
#include <hip/hip_runtime.h>
#include <hip/hip_bf16.h>

typedef __attribute__((ext_vector_type(16))) _Float16 v16h;
typedef __attribute__((ext_vector_type(8)))  _Float16 v8h;
typedef __attribute__((ext_vector_type(8)))  float    v8f;
typedef __attribute__((ext_vector_type(4)))  unsigned int u32x4;
typedef __attribute__((ext_vector_type(8)))  int          i32x8;
typedef __attribute__((ext_vector_type(4)))  int          i32x4;

#define BB 32
#define TT 2000
#define INDIM 118
#define HIDD 256
#define OUTDIM 118
#define MROWS (BB * TT)        // 64000

// LDS A-tile: 128 rows x (32 data + 8 pad) halves = 10240 B per buffer.
#define A_ROW_STRIDE 40
#define A_TILE_BYTES (128 * A_ROW_STRIDE * 2)

// ---------------------------------------------------------------------------
// Pack kernels. A-side: fp32 -> fp16 row-major, K zero-padded to mult of 32.
// B-side (weights): fp32 -> fp16 TRANSPOSED to [N][K] (K contiguous) so WMMA
// B fragments are two contiguous 16B runs per lane, loaded straight from
// global (weights are <=0.5MB -> permanently L2-resident on a 192MB L2).
// ---------------------------------------------------------------------------
__global__ void pack_x_f16(const float* __restrict__ src, _Float16* __restrict__ dst) {
    long i = (long)blockIdx.x * blockDim.x + threadIdx.x;
    long total = (long)MROWS * 128;
    if (i >= total) return;
    long row = i >> 7;
    int  col = (int)(i & 127);
    float v = (col < INDIM) ? src[row * INDIM + col] : 0.0f;
    dst[i] = (_Float16)v;
}

__global__ void pack_w0t_f16(const float* __restrict__ src, _Float16* __restrict__ dst) {
    // src: [118, 1024] row-major -> dst: [1024][128] (n-major, k contiguous)
    long i = (long)blockIdx.x * blockDim.x + threadIdx.x;
    long total = 1024L * 128;
    if (i >= total) return;
    int n = (int)(i >> 7);
    int k = (int)(i & 127);
    float v = (k < INDIM) ? src[(size_t)k * 1024 + n] : 0.0f;
    dst[i] = (_Float16)v;
}

__global__ void pack_w1t_f16(const float* __restrict__ src, _Float16* __restrict__ dst) {
    // src: [256, 768] -> dst: [768][256]
    long i = (long)blockIdx.x * blockDim.x + threadIdx.x;
    long total = 768L * 256;
    if (i >= total) return;
    int n = (int)(i >> 8);
    int k = (int)(i & 255);
    dst[i] = (_Float16)src[(size_t)k * 768 + n];
}

__global__ void pack_wot_f16(const float* __restrict__ src, _Float16* __restrict__ dst) {
    // src: [256, 118] -> dst: [128][256], n zero-padded
    long i = (long)blockIdx.x * blockDim.x + threadIdx.x;
    long total = 128L * 256;
    if (i >= total) return;
    int n = (int)(i >> 8);
    int k = (int)(i & 255);
    float v = (n < OUTDIM) ? src[(size_t)k * OUTDIM + n] : 0.0f;
    dst[i] = (_Float16)v;
}

// ---------------------------------------------------------------------------
// TDM descriptor build + issue: load a 128x32 fp16 tile (row stride K elems)
// from global into LDS at ldsByteAddr, inserting 16B of padding after every
// 64B row so LDS rows land on the 40-half conflict-free stride.
// D# layout per CDNA5 ISA.
// Toolchain here is the 6-arg clang-23 builtin:
//   (u32x4 g0, i32x8 g1, i32x4 g2, i32x4 g3, i32x8 g4, i32 cpol)
// ---------------------------------------------------------------------------
__device__ __forceinline__ void tdm_load_a_tile(const _Float16* gsrc,
                                                unsigned ldsByteAddr, int K)
{
    unsigned long long ga = (unsigned long long)(const void*)gsrc;
    u32x4 g0;
    g0[0] = 1u;                                        // count=1 (valid user D#)
    g0[1] = ldsByteAddr;                               // lds_addr (bytes)
    g0[2] = (unsigned)(ga & 0xFFFFFFFFu);              // global_addr[31:0]
    g0[3] = (unsigned)((ga >> 32) & 0x01FFFFFFu)       // global_addr[56:32]
          | 0x80000000u;                               // type=2 ("image")
    i32x8 g1;
    g1[0] = (1 << 16)      // data_size = 2 bytes
          | (1 << 20)      // pad_enable
          | (3 << 22)      // pad_interval: 16 DWORDs (64B = one 32-half row)
          | (3 << 25);     // pad_amount: 4 DWORDs (16B = 8 halves)
    g1[1] = (32 << 16);                                // tensor_dim0 = 32 (lo16)
    g1[2] = (128 << 16);                               // dim0 hi16=0 | tensor_dim1=128 lo16
    g1[3] = (32 << 16);                                // dim1 hi16=0 | tile_dim0 = 32
    g1[4] = 128;                                       // tile_dim1 = 128, tile_dim2 = 0
    g1[5] = K;                                         // tensor_dim0_stride lo32 (elems)
    g1[6] = 0;
    g1[7] = 0;
    i32x4 gz4 = {0, 0, 0, 0};                          // groups 2/3: 2-D tile
    i32x8 gz8 = {0, 0, 0, 0, 0, 0, 0, 0};              // extra group (6-arg form)
    __builtin_amdgcn_tensor_load_to_lds(g0, g1, gz4, gz4, gz8, 0);
}

// ---------------------------------------------------------------------------
// WMMA GEMM: C[M,N] = A[M,K] * Bt[N,K]^T (+bias), fp16 in / fp32 accumulate.
// Block = 256 threads = 8 waves, block tile 128(M) x 128(N).
// Wave tile 32 x 64: 2 A-frags x 4 B-frags -> 8 v_wmma per K-step (K step 32).
// A tile is DMA'd into LDS by the Tensor Data Mover (wave 0 issues
// tensor_load_to_lds, double-buffered, s_wait_tensorcnt overlap); B fragments
// come straight from global (transposed, L2-resident weights) as 16B loads.
// CDNA5 16-bit frag layout: lane%16 = row(A)/col(B); lane/16 picks K-halves
// {0-7,16-23} vs {8-15,24-31}. C/D: lane%16 = N, VGPR r -> M = r+8*(lane/16).
// ---------------------------------------------------------------------------
__global__ __launch_bounds__(256)
void gemm_wmma_f16(const _Float16* __restrict__ A, const _Float16* __restrict__ Bt,
                   float* __restrict__ C, int K, int ldc, int nstore,
                   const float* __restrict__ bias)
{
    __shared__ _Float16 As[2][128 * A_ROW_STRIDE];   // double-buffered TDM target

    const int tid  = threadIdx.x;
    const int wave = tid >> 5;
    const int lane = tid & 31;
    const int row  = lane & 15;
    const int half = lane >> 4;
    const int mw   = wave & 3;          // 32-row strip within block
    const int nw   = wave >> 2;         // 64-col strip within block

    const int m0 = blockIdx.x * 128;
    const int n0 = blockIdx.y * 128 + nw * 64;

    const unsigned ldsBase = (unsigned)(size_t)(void*)&As[0][0];

    v8f acc[2][4] = {};

    // prologue: DMA tile k0=0 into buffer 0
    if (wave == 0) {
        tdm_load_a_tile(A + (size_t)m0 * K, ldsBase, K);
    }

    int cur = 0;
    for (int k0 = 0; k0 < K; k0 += 32) {
        // all waves done reading buffer cur^1 from the previous iteration
        __syncthreads();
        if (wave == 0) {
            if (k0 + 32 < K) {
                // kick next tile into the other buffer, then wait for current
                tdm_load_a_tile(A + (size_t)m0 * K + (k0 + 32),
                                ldsBase + (unsigned)((cur ^ 1) * A_TILE_BYTES), K);
                __builtin_amdgcn_s_wait_tensorcnt(1);
            } else {
                __builtin_amdgcn_s_wait_tensorcnt(0);
            }
        }
        __syncthreads();   // buffer cur now resident for the whole block

        const _Float16* Ab = &As[cur][0];

        // A fragments: two 16B LDS loads each
        v16h afrag[2];
        #pragma unroll
        for (int p = 0; p < 2; ++p) {
            int r = mw * 32 + p * 16 + row;
            v8h lo = *reinterpret_cast<const v8h*>(&Ab[r * A_ROW_STRIDE + 8 * half]);
            v8h hi = *reinterpret_cast<const v8h*>(&Ab[r * A_ROW_STRIDE + 16 + 8 * half]);
            afrag[p] = __builtin_shufflevector(lo, hi,
                0, 1, 2, 3, 4, 5, 6, 7, 8, 9, 10, 11, 12, 13, 14, 15);
        }

        // B fragments straight from global (transposed, L2-resident weights)
        #pragma unroll
        for (int s = 0; s < 4; ++s) {
            int gn = n0 + s * 16 + row;
            const _Float16* bp = &Bt[(size_t)gn * K + k0 + 8 * half];
            v8h lo = *reinterpret_cast<const v8h*>(bp);
            v8h hi = *reinterpret_cast<const v8h*>(bp + 16);
            v16h bfrag = __builtin_shufflevector(lo, hi,
                0, 1, 2, 3, 4, 5, 6, 7, 8, 9, 10, 11, 12, 13, 14, 15);
            #pragma unroll
            for (int p = 0; p < 2; ++p) {
                acc[p][s] = __builtin_amdgcn_wmma_f32_16x16x32_f16(
                                false, afrag[p], false, bfrag,
                                (short)0, acc[p][s], false, false);
            }
        }
        cur ^= 1;
    }

    // epilogue: lane%16 = n, VGPR r -> m = r + 8*half (per 16x16 tile)
    #pragma unroll
    for (int s = 0; s < 4; ++s) {
        int gn = n0 + s * 16 + row;
        if (gn < nstore) {
            float bval = bias ? bias[gn] : 0.0f;
            #pragma unroll
            for (int p = 0; p < 2; ++p) {
                #pragma unroll
                for (int r = 0; r < 8; ++r) {
                    int gm = m0 + mw * 32 + p * 16 + half * 8 + r;
                    C[(size_t)gm * ldc + gn] = acc[p][s][r] + bval;
                }
            }
        }
    }
}

// ---------------------------------------------------------------------------
// SRU recurrence, layer 0 (k=4). One thread per (batch, channel), d
// contiguous -> coalesced streaming of U (262MB) at HBM rate. Prefetch the
// t+8 row (global_prefetch_b8) while the serial c-chain executes.
// ---------------------------------------------------------------------------
__global__ __launch_bounds__(256)
void sru_recur_k4(const float* __restrict__ U, const float* __restrict__ bvec,
                  float* __restrict__ hf, _Float16* __restrict__ hh)
{
    int tid = blockIdx.x * blockDim.x + threadIdx.x;   // 0..8191
    int b   = tid >> 8;
    int d   = tid & 255;
    float bf = bvec[d];
    float br = bvec[256 + d];
    float c  = 0.0f;
    size_t base = (size_t)b * TT;
    for (int t = 0; t < TT; ++t) {
        const float* u = U + (base + t) * 1024;
        __builtin_prefetch(u + 8 * 1024 + d, 0, 1);
        float xt  = u[d];
        float f   = 1.0f / (1.0f + __expf(-(u[256 + d] + bf)));
        float r   = 1.0f / (1.0f + __expf(-(u[512 + d] + br)));
        float xhw = u[768 + d];
        c = f * (c - xt) + xt;                 // f*c + (1-f)*xt
        float h = r * tanhf(c) + (1.0f - r) * xhw;
        size_t o = (base + t) * 256 + d;
        hf[o] = h;
        hh[o] = (_Float16)h;
    }
}

// ---------------------------------------------------------------------------
// SRU recurrence, layer 1 (k=3: x_hw is the layer input h1).
// ---------------------------------------------------------------------------
__global__ __launch_bounds__(256)
void sru_recur_k3(const float* __restrict__ U, const float* __restrict__ bvec,
                  const float* __restrict__ xin, _Float16* __restrict__ hh)
{
    int tid = blockIdx.x * blockDim.x + threadIdx.x;
    int b   = tid >> 8;
    int d   = tid & 255;
    float bf = bvec[d];
    float br = bvec[256 + d];
    float c  = 0.0f;
    size_t base = (size_t)b * TT;
    for (int t = 0; t < TT; ++t) {
        const float* u = U + (base + t) * 768;
        __builtin_prefetch(u + 8 * 768 + d, 0, 1);
        size_t o = (base + t) * 256 + d;
        float xt  = u[d];
        float f   = 1.0f / (1.0f + __expf(-(u[256 + d] + bf)));
        float r   = 1.0f / (1.0f + __expf(-(u[512 + d] + br)));
        float xhw = xin[o];
        c = f * (c - xt) + xt;
        float h = r * tanhf(c) + (1.0f - r) * xhw;
        hh[o] = (_Float16)h;
    }
}

// ---------------------------------------------------------------------------
// Orchestration. Workspace (~410 MB): U buffer reused between layers.
// lengths (d_in[1]) is unused by the reference -> ignored.
// ---------------------------------------------------------------------------
extern "C" void kernel_launch(void* const* d_in, const int* in_sizes, int n_in,
                              void* d_out, int out_size, void* d_ws, size_t ws_size,
                              hipStream_t stream)
{
    (void)in_sizes; (void)n_in; (void)out_size; (void)ws_size;

    const float* seq  = (const float*)d_in[0];
    const float* W0   = (const float*)d_in[2];
    const float* b0   = (const float*)d_in[3];
    const float* W1   = (const float*)d_in[4];
    const float* b1   = (const float*)d_in[5];
    const float* Wout = (const float*)d_in[6];
    const float* bout = (const float*)d_in[7];
    float* out = (float*)d_out;

    char* ws = (char*)d_ws;
    auto carve = [&](size_t bytes) -> char* {
        char* p = ws;
        ws += (bytes + 255) & ~(size_t)255;
        return p;
    };
    float*    U    = (float*)   carve((size_t)MROWS * 1024 * 4);  // U0 then U1
    float*    h1f  = (float*)   carve((size_t)MROWS * 256 * 4);
    _Float16* h1h  = (_Float16*)carve((size_t)MROWS * 256 * 2);
    _Float16* h2h  = (_Float16*)carve((size_t)MROWS * 256 * 2);
    _Float16* Xh   = (_Float16*)carve((size_t)MROWS * 128 * 2);
    _Float16* W0t  = (_Float16*)carve(1024ull * 128 * 2);
    _Float16* W1t  = (_Float16*)carve(768ull * 256 * 2);
    _Float16* Wot  = (_Float16*)carve(128ull * 256 * 2);

    // --- pack fp32 -> fp16 (A padded row-major; weights transposed [N][K])
    pack_x_f16  <<<(MROWS * 128 + 255) / 256, 256, 0, stream>>>(seq, Xh);
    pack_w0t_f16<<<(1024 * 128 + 255) / 256, 256, 0, stream>>>(W0, W0t);
    pack_w1t_f16<<<(768 * 256 + 255) / 256, 256, 0, stream>>>(W1, W1t);
    pack_wot_f16<<<(128 * 256 + 255) / 256, 256, 0, stream>>>(Wout, Wot);

    // --- layer 0: U0 = X[64000,128] @ W0[128,1024]
    gemm_wmma_f16<<<dim3(MROWS / 128, 1024 / 128), 256, 0, stream>>>(
        Xh, W0t, U, /*K=*/128, /*ldc=*/1024, /*nstore=*/1024, nullptr);
    sru_recur_k4<<<(BB * HIDD) / 256, 256, 0, stream>>>(U, b0, h1f, h1h);

    // --- layer 1: U1 = h1[64000,256] @ W1[256,768] (reuses U buffer)
    gemm_wmma_f16<<<dim3(MROWS / 128, 768 / 128), 256, 0, stream>>>(
        h1h, W1t, U, /*K=*/256, /*ldc=*/768, /*nstore=*/768, nullptr);
    sru_recur_k3<<<(BB * HIDD) / 256, 256, 0, stream>>>(U, b1, h1f, h2h);

    // --- output: out = h2[64000,256] @ Wout[256,128pad] + bout, clip N to 118
    gemm_wmma_f16<<<dim3(MROWS / 128, 1), 256, 0, stream>>>(
        h2h, Wot, out, /*K=*/256, /*ldc=*/OUTDIM, /*nstore=*/OUTDIM, bout);
}
